// PAdicGVP_19567871001257
// MI455X (gfx1250) — matrix operand order repaired and checked by reading, not compiled, not run on hardware
//
#include <hip/hip_runtime.h>
#include <hip/hip_bf16.h>

#define BB 8
#define NN 1024
#define SS 64
#define VCC 8
#define EE 64
#define HH 4
#define HDD 16

typedef __attribute__((ext_vector_type(16))) _Float16 v16h;
typedef __attribute__((ext_vector_type(8)))  _Float16 v8h;
typedef __attribute__((ext_vector_type(8)))  float    v8f;

__device__ __forceinline__ float sigmoidf_(float x) { return 1.0f / (1.0f + __expf(-x)); }

__device__ __forceinline__ int padic_val(float d) {
    int di = (int)(fabsf(d) * 1000.0f);
    if (di <= 0) return 0;
    int c = 0;
    c += (di % 3)     == 0;
    c += (di % 9)     == 0;
    c += (di % 27)    == 0;
    c += (di % 81)    == 0;
    c += (di % 243)   == 0;
    c += (di % 729)   == 0;
    c += (di % 2187)  == 0;
    c += (di % 6561)  == 0;
    c += (di % 19683) == 0;
    c += (di % 59049) == 0;
    return c;
}

// ---------------- Stage 1: pd (B,N,N) + row means -------------------------
__global__ __launch_bounds__(256)
void pd_kernel(const float* __restrict__ coords, float* __restrict__ pd,
               float* __restrict__ rowmean) {
    int row = blockIdx.x;              // b*N + i
    int b = row >> 10;
    const float* ci = coords + (size_t)row * 3;
    float cx = ci[0], cy = ci[1], cz = ci[2];
    const float* cb = coords + (size_t)b * NN * 3;
    float* prow = pd + (size_t)row * NN;

    float lsum = 0.0f;
    for (int j = threadIdx.x; j < NN; j += blockDim.x) {
        float dx = cx - cb[j * 3 + 0];
        float dy = cy - cb[j * 3 + 1];
        float dz = cz - cb[j * 3 + 2];
        int c = padic_val(dx) + padic_val(dy) + padic_val(dz);
        float val = (float)c * (1.0f / 3.0f);
        float p = exp2f(-1.5849625007211562f * val);   // 3^(-val)
        prow[j] = p;
        lsum += p;
    }
    // block reduce (wave32)
    #pragma unroll
    for (int off = 16; off > 0; off >>= 1) lsum += __shfl_down(lsum, off, 32);
    __shared__ float ws_[8];
    int wid = threadIdx.x >> 5, wl = threadIdx.x & 31;
    if (wl == 0) ws_[wid] = lsum;
    __syncthreads();
    if (threadIdx.x == 0) {
        float t = 0.0f;
        #pragma unroll
        for (int w = 0; w < 8; w++) t += ws_[w];
        rowmean[row] = t * (1.0f / (float)NN);
    }
}

// ---------------- Stage 2: p-adic encoder + residual into sbuf ------------
__global__ __launch_bounds__(64)
void enc_kernel(const float* __restrict__ s, const float* __restrict__ rowmean,
                const float* __restrict__ w1, const float* __restrict__ b1,
                const float* __restrict__ w2, const float* __restrict__ b2,
                float* __restrict__ sbuf) {
    int node = blockIdx.x;
    int e = threadIdx.x;
    float x = rowmean[node];
    float pf = b2[e];
    #pragma unroll
    for (int t = 0; t < 16; t++) {
        float h = x * w1[t] + b1[t];
        h = h * sigmoidf_(h);                 // SiLU
        pf += h * w2[e * 16 + t];
    }
    size_t idx = (size_t)node * EE + e;
    sbuf[idx] = s[idx] + pf;
}

// ---------------- Stage 3: 3 GVP layers (in-place on sbuf, v -> d_out) ----
__global__ __launch_bounds__(64)
void gvp_kernel(const float* __restrict__ vin,
                const float* __restrict__ scal_w, const float* __restrict__ scal_b,
                const float* __restrict__ vec_w,  const float* __restrict__ gate_w,
                const float* __restrict__ gate_b,
                float* __restrict__ sbuf, float* __restrict__ vout) {
    __shared__ float s_cur[64], v_cur[24], vn[8], snew[64], vnew[24], gate[8];
    int node = blockIdx.x;
    int t = threadIdx.x;
    s_cur[t] = sbuf[(size_t)node * EE + t];
    if (t < 24) v_cur[t] = vin[(size_t)node * 24 + t];
    __syncthreads();

    for (int i = 0; i < 3; i++) {
        bool act = (i < 2);
        if (t < 8) {
            float a = v_cur[t * 3], b = v_cur[t * 3 + 1], c = v_cur[t * 3 + 2];
            vn[t] = sqrtf(a * a + b * b + c * c);
        }
        __syncthreads();
        // s_new = sc @ scal_w^T + b
        {
            const float* sw = scal_w + (size_t)i * 64 * 72 + (size_t)t * 72;
            float acc = scal_b[i * 64 + t];
            #pragma unroll 8
            for (int k = 0; k < 64; k++) acc += sw[k] * s_cur[k];
            #pragma unroll
            for (int k = 0; k < 8; k++) acc += sw[64 + k] * vn[k];
            if (act) acc = acc * sigmoidf_(acc);     // SiLU
            snew[t] = acc;
        }
        if (t < 8) {
            const float* gw = gate_w + (size_t)i * 8 * 72 + (size_t)t * 72;
            float g = gate_b[i * 8 + t];
            #pragma unroll 8
            for (int k = 0; k < 64; k++) g += gw[k] * s_cur[k];
            #pragma unroll
            for (int k = 0; k < 8; k++) g += gw[64 + k] * vn[k];
            gate[t] = sigmoidf_(g);
        }
        if (t < 24) {
            int o = t / 3, x = t % 3;
            const float* vw = vec_w + (size_t)i * 64 + o * 8;
            float a = 0.0f;
            #pragma unroll
            for (int c = 0; c < 8; c++) a += vw[c] * v_cur[c * 3 + x];
            vnew[t] = a;
        }
        __syncthreads();
        if (t < 24) {
            int o = t / 3;
            float f = gate[o];
            if (act) {
                float a = vnew[o * 3], b = vnew[o * 3 + 1], c = vnew[o * 3 + 2];
                f *= sigmoidf_(sqrtf(a * a + b * b + c * c));
            }
            float nv = vnew[t] * f;
            v_cur[t] = (i > 0) ? (v_cur[t] + nv) : nv;
        }
        float ns = snew[t];
        s_cur[t] = (i > 0) ? (s_cur[t] + ns) : ns;
        __syncthreads();
    }
    sbuf[(size_t)node * EE + t] = s_cur[t];
    if (t < 24) vout[(size_t)node * 24 + t] = v_cur[t];
}

// ---------------- Stage 4: QKV projection to f16 (V transposed) -----------
__global__ __launch_bounds__(64)
void qkv_kernel(const float* __restrict__ sbuf, const float* __restrict__ inw,
                const float* __restrict__ inb,
                _Float16* __restrict__ qh, _Float16* __restrict__ kh,
                _Float16* __restrict__ vth) {
    __shared__ float sl[64];
    int node = blockIdx.x;
    int e = threadIdx.x;
    int b = node >> 10, n = node & 1023;
    sl[e] = sbuf[(size_t)node * EE + e];
    __syncthreads();
    float q = inb[e], k = inb[64 + e], v = inb[128 + e];
    const float* wq = inw + (size_t)e * 64;
    const float* wk = inw + (size_t)(64 + e) * 64;
    const float* wv = inw + (size_t)(128 + e) * 64;
    #pragma unroll 8
    for (int kk = 0; kk < 64; kk++) {
        float x = sl[kk];
        q += wq[kk] * x; k += wk[kk] * x; v += wv[kk] * x;
    }
    int h = e >> 4, d = e & 15;
    size_t qi = ((((size_t)b * HH + h) * NN + n) * HDD + d);
    qh[qi] = (_Float16)q;
    kh[qi] = (_Float16)k;
    vth[(((size_t)b * HH + h) * HDD + d) * NN + n] = (_Float16)v;
}

// ---------------- Stage 5: flash attention with WMMA (wave32) -------------
// 4 independent waves per block; each wave owns one (b,h,16-query tile).
__global__ __launch_bounds__(128)
void attn_kernel(const _Float16* __restrict__ qh, const _Float16* __restrict__ kh,
                 const _Float16* __restrict__ vth, const float* __restrict__ pd,
                 float* __restrict__ obuf) {
    int tile = blockIdx.x * 4 + (threadIdx.x >> 5);   // 0..2047
    int qt = tile & 63;
    int h  = (tile >> 6) & 3;
    int b  = tile >> 8;
    int lane = threadIdx.x & 31;
    int n  = lane & 15;          // query column within tile / matrix row index
    int hi = lane >> 4;          // lane half
    int kb = hi * 8;             // K-offset owned by this lane half
    int q0 = qt * 16;

    // B operand of WMMA-1: Q^T  (lane half h: halfs0-7 = Q[q0+n, 8h..8h+7]; halfs8-15 = 0 pad)
    const _Float16* qp = qh + ((((size_t)b * HH + h) * NN + (q0 + n)) * HDD);
    v16h bq;
    #pragma unroll
    for (int r = 0; r < 16; r++) {
        bq[r] = (r < 8) ? qp[kb + r] : (_Float16)0.0f;
    }

    float m_run = -INFINITY, s_run = 0.0f;
    v8f acc = {0.f, 0.f, 0.f, 0.f, 0.f, 0.f, 0.f, 0.f};
    v8f zc  = {0.f, 0.f, 0.f, 0.f, 0.f, 0.f, 0.f, 0.f};

    const float*    pdrow = pd  + (((size_t)b * NN + (q0 + n)) * NN);
    const _Float16* kbase = kh  + ((((size_t)b * HH + h) * NN) * HDD);
    const _Float16* vbase = vth + ((((size_t)b * HH + h) * HDD + n) * NN);  // row d = n

    #pragma unroll 2
    for (int t = 0; t < 64; t++) {
        // Prefetch K/V tiles two iterations ahead (global_prefetch_b8)
        {
            int tn = (t + 2 < 64) ? (t + 2) : 63;
            __builtin_prefetch(kbase + ((size_t)(tn * 16 + n)) * HDD + kb, 0, 1);
            __builtin_prefetch(vbase + tn * 16 + kb, 0, 1);
        }
        // A = K tile (16 keys x 16 head-dim, padded to K=32)
        const _Float16* kp = kbase + ((size_t)(t * 16 + n)) * HDD + kb;
        v16h ak;
        #pragma unroll
        for (int r = 0; r < 8; r++) { ak[r] = kp[r]; ak[8 + r] = (_Float16)0.0f; }

        // S^T tile: rows = keys, cols = queries
        v8f st = __builtin_amdgcn_wmma_f32_16x16x32_f16(false, ak, false, bq,
                                                        (short)0, zc, false, false);
        // bias + online softmax (query i = q0+n ; key j = t*16 + kb + r)
        const float* pdp = pdrow + t * 16 + kb;
        float p[8];
        float tm = -INFINITY;
        #pragma unroll
        for (int r = 0; r < 8; r++) {
            float sv = st[r] * 0.25f - pdp[r];
            p[r] = sv;
            tm = fmaxf(tm, sv);
        }
        tm = fmaxf(tm, __shfl_xor(tm, 16, 32));
        float m_new = fmaxf(m_run, tm);
        float corr  = __expf(m_run - m_new);
        float sp = 0.0f;
        #pragma unroll
        for (int r = 0; r < 8; r++) { p[r] = __expf(p[r] - m_new); sp += p[r]; }
        s_run = s_run * corr + (sp + __shfl_xor(sp, 16, 32));
        m_run = m_new;
        #pragma unroll
        for (int r = 0; r < 8; r++) acc[r] *= corr;

        // Pack own 8 probs to f16 BEFORE the lane exchange: 4 dword shuffles
        // instead of 8 float shuffles (halves the ds_bpermute traffic).
        v8h pown;
        #pragma unroll
        for (int r = 0; r < 8; r++) pown[r] = (_Float16)p[r];
        int pu[4], pq[4];
        __builtin_memcpy(pu, &pown, 16);
        #pragma unroll
        for (int w = 0; w < 4; w++) pq[w] = __shfl_xor(pu[w], 16, 32);

        // B operand of WMMA-2: P^T (halfs0-7 = own p (j=kb..kb+7),
        // halfs8-15 zero = K pad)
        int bpraw[8];
        #pragma unroll
        for (int w = 0; w < 4; w++) {
            bpraw[w]     = pu[w];
            bpraw[4 + w] = 0;
        }
        v16h bp;
        __builtin_memcpy(&bp, bpraw, 32);

        // A = V^T tile (rows = head-dim d, K = keys, padded to 32)
        const _Float16* vp = vbase + t * 16 + kb;
        v16h av;
        #pragma unroll
        for (int r = 0; r < 8; r++) { av[r] = vp[r]; av[8 + r] = (_Float16)0.0f; }

        acc = __builtin_amdgcn_wmma_f32_16x16x32_f16(false, av, false, bp,
                                                     (short)0, acc, false, false);
    }

    float inv = 1.0f / s_run;
    // D layout: lane holds column i=q0+n, rows d = r + 8*hi  -> o[b, q0+n, h*16+d]
    float* op = obuf + (((size_t)b * NN + (q0 + n)) * EE + h * HDD + hi * 8);
    #pragma unroll
    for (int r = 0; r < 8; r++) op[r] = acc[r] * inv;
}

// ---------------- Stage 6: output projection + residual -------------------
__global__ __launch_bounds__(64)
void outproj_kernel(const float* __restrict__ sbuf, const float* __restrict__ obuf,
                    const float* __restrict__ ow, const float* __restrict__ ob,
                    float* __restrict__ sout) {
    __shared__ float ol[64];
    int node = blockIdx.x;
    int e = threadIdx.x;
    ol[e] = obuf[(size_t)node * EE + e];
    __syncthreads();
    float a = ob[e];
    const float* w = ow + (size_t)e * 64;
    #pragma unroll 8
    for (int k = 0; k < 64; k++) a += w[k] * ol[k];
    sout[(size_t)node * EE + e] = sbuf[(size_t)node * EE + e] + a;
}

extern "C" void kernel_launch(void* const* d_in, const int* in_sizes, int n_in,
                              void* d_out, int out_size, void* d_ws, size_t ws_size,
                              hipStream_t stream) {
    (void)in_sizes; (void)n_in; (void)out_size; (void)ws_size;

    const float* s       = (const float*)d_in[0];
    const float* v       = (const float*)d_in[1];
    const float* coords  = (const float*)d_in[2];
    const float* enc_w1  = (const float*)d_in[3];
    const float* enc_b1  = (const float*)d_in[4];
    const float* enc_w2  = (const float*)d_in[5];
    const float* enc_b2  = (const float*)d_in[6];
    const float* scal_w  = (const float*)d_in[7];
    const float* scal_b  = (const float*)d_in[8];
    const float* vec_w   = (const float*)d_in[9];
    const float* gate_w  = (const float*)d_in[10];
    const float* gate_b  = (const float*)d_in[11];
    const float* in_w    = (const float*)d_in[12];
    const float* in_b    = (const float*)d_in[13];
    const float* out_w   = (const float*)d_in[14];
    const float* out_b   = (const float*)d_in[15];

    float* out   = (float*)d_out;
    float* s_out = out;                       // (B,N,E)      524288
    float* v_out = out + 524288;              // (B,N,VC,3)   196608
    float* pd    = out + 720896;              // (B,N,N)      8388608

    float* ws        = (float*)d_ws;
    float* rowmean   = ws;                    //  8192 f
    float* sbuf      = rowmean + 8192;        //  524288 f
    float* obuf      = sbuf + 524288;         //  524288 f
    _Float16* qh     = (_Float16*)(obuf + 524288);  // 524288 h
    _Float16* kh     = qh + 524288;
    _Float16* vth    = kh + 524288;

    const int nodes = BB * NN;                // 8192

    pd_kernel<<<nodes, 256, 0, stream>>>(coords, pd, rowmean);
    enc_kernel<<<nodes, 64, 0, stream>>>(s, rowmean, enc_w1, enc_b1, enc_w2, enc_b2, sbuf);
    gvp_kernel<<<nodes, 64, 0, stream>>>(v, scal_w, scal_b, vec_w, gate_w, gate_b, sbuf, v_out);
    qkv_kernel<<<nodes, 64, 0, stream>>>(sbuf, in_w, in_b, qh, kh, vth);
    attn_kernel<<<512, 128, 0, stream>>>(qh, kh, vth, pd, obuf);
    outproj_kernel<<<nodes, 64, 0, stream>>>(sbuf, obuf, out_w, out_b, s_out);
}
